// GCN2_7413113552905
// MI455X (gfx1250) — matrix operand order repaired and checked by reading, not compiled, not run on hardware
//
#include <hip/hip_runtime.h>
#include <hip/hip_bf16.h>
#include <math.h>

// ---------------------------------------------------------------------------
// GENConv x3 + BN + softmax-aggregation + mean-pool for MI455X (gfx1250).
// Dense linears -> v_wmma_f32_16x16x32_f16 (f32 accum), A fragments loaded as
// b128 vectors + VALU cvt, B fragments pre-packed to register layout so each
// lane issues two contiguous b128 loads per K-step.  Edge softmax aggregation
// is 2 fused passes (atomicMax on positive-float bits, then one pass that
// accumulates both Sum(e^x) and Sum(msg*e^x)) so the E x O message tensor is
// never materialized in HBM.
// ---------------------------------------------------------------------------

typedef __attribute__((ext_vector_type(16))) _Float16 v16h;
typedef __attribute__((ext_vector_type(8)))  _Float16 v8h;
typedef __attribute__((ext_vector_type(8)))  float    v8f;

#define EPS_MSG 1e-7f
#define EPS_SMX 1e-16f
#define EPS_BN  1e-5f
#define NUM_GRAPHS 64

// ----------------------------- fill ---------------------------------------
__global__ void k_fill(float* __restrict__ p, float v, long long n) {
  long long t    = (long long)blockIdx.x * blockDim.x + threadIdx.x;
  long long step = (long long)gridDim.x * blockDim.x;
  for (; t < n; t += step) p[t] = v;
}

// ------ pack W[K,N] (f32, row major) into per-lane B-fragment f16 layout ----
// out[((kt*(N/16) + nt)*32 + lane)*16 + j] = (f16) W[(kt*32 + (lane>>4)*8 + (j&7) + (j>>3)*16)*N
//                                                   + nt*16 + (lane&15)]
__global__ void k_pack_b(const float* __restrict__ W, _Float16* __restrict__ out,
                         int K, int N) {
  int total = (K / 32) * (N / 16) * 32 * 16;
  int t = blockIdx.x * blockDim.x + threadIdx.x;
  if (t >= total) return;
  int j    = t & 15;
  int lane = (t >> 4) & 31;
  int tile = t >> 9;
  int Ntiles = N / 16;
  int nt = tile % Ntiles;
  int kt = tile / Ntiles;
  int kb   = kt * 32 + (lane >> 4) * 8 + (j & 7) + (j >> 3) * 16;
  int ncol = nt * 16 + (lane & 15);
  out[t] = (_Float16)W[(long long)kb * N + ncol];
}

// ------------- WMMA GEMM: C = A[M,K] @ W[K,N] + bias  (K % 32 == 0) --------
// wave32: one wave owns a 16x16 output tile; 4 waves / block.
__global__ __launch_bounds__(128)
void k_gemm_wmma(const float* __restrict__ A, const _Float16* __restrict__ Bp,
                 const float* __restrict__ bias, float* __restrict__ C,
                 int M, int K, int N, int relu) {
  const int lane = threadIdx.x;                   // 0..31
  const int wave = threadIdx.y;                   // 0..3
  const int nt   = blockIdx.x;                    // N-tile index
  const int n0   = nt * 16;
  const int m0   = (blockIdx.y * 4 + wave) * 16;
  if (m0 >= M) return;                            // wave-uniform

  const int       half   = lane >> 4;             // 0 | 1
  const int       Ntiles = N >> 4;
  const long long mrow   = m0 + (lane & 15);
  const int       ncol   = n0 + (lane & 15);

  const float*    arow = A + mrow * (long long)K;
  v8f acc = {};
  for (int k0 = 0, kt = 0; k0 < K; k0 += 32, ++kt) {
    // A fragment: two contiguous 8-float runs -> 4 x global_load_b128 + cvt
    const float4* ap = (const float4*)(arow + k0 + half * 8);
    float4 a0 = ap[0], a1 = ap[1];                // K = k0 + half*8 + 0..7
    float4 a2 = ap[4], a3 = ap[5];                // K = k0 + 16 + half*8 + 0..7
    v16h a;
    a[0]=(_Float16)a0.x;  a[1]=(_Float16)a0.y;  a[2]=(_Float16)a0.z;  a[3]=(_Float16)a0.w;
    a[4]=(_Float16)a1.x;  a[5]=(_Float16)a1.y;  a[6]=(_Float16)a1.z;  a[7]=(_Float16)a1.w;
    a[8]=(_Float16)a2.x;  a[9]=(_Float16)a2.y;  a[10]=(_Float16)a2.z; a[11]=(_Float16)a2.w;
    a[12]=(_Float16)a3.x; a[13]=(_Float16)a3.y; a[14]=(_Float16)a3.z; a[15]=(_Float16)a3.w;

    // B fragment: pre-packed -> two contiguous b128 loads
    const v8h* bp = (const v8h*)(Bp + ((long long)(kt * Ntiles + nt) * 32 + lane) * 16);
    v8h blo = bp[0], bhi = bp[1];
    v16h b = __builtin_shufflevector(blo, bhi, 0,1,2,3,4,5,6,7,8,9,10,11,12,13,14,15);

    acc = __builtin_amdgcn_wmma_f32_16x16x32_f16(false, a, false, b,
                                                 (short)0, acc, false, false);
  }
#pragma unroll
  for (int r = 0; r < 8; ++r) {                   // C/D: VGPR r -> row r (+8 hi)
    long long row = m0 + r + half * 8;
    float v = acc[r] + bias[ncol];
    if (relu) v = fmaxf(v, 0.0f);
    C[row * (long long)N + ncol] = v;
  }
}

// --------------------- edge pass 1: segment max ----------------------------
// msg = relu(h_src[src] + edge_attr @ We + be) + 1e-7  (strictly > 0, so
// uint-bit atomicMax is order preserving; segments init'd to 0.0f bits).
__global__ void k_edge_max(const float* __restrict__ ea,   // [E,16]
                           const float* __restrict__ We,   // [16,O]
                           const float* __restrict__ be,   // [O]
                           const float* __restrict__ hsrc, // [N,O]
                           const long long* __restrict__ src,
                           const long long* __restrict__ dst,
                           unsigned int* __restrict__ mbits, // [N,O]
                           long long E, int shO) {
  const int       O     = 1 << shO;
  const long long total = E << shO;
  const long long step  = (long long)gridDim.x * blockDim.x;
  for (long long t = (long long)blockIdx.x * blockDim.x + threadIdx.x;
       t < total; t += step) {
    long long e = t >> shO;
    int       c = (int)(t & (O - 1));
    long long s = src[e], d = dst[e];
    const float* row = ea + e * 16;
    __builtin_prefetch(ea + (e + 256) * 16, 0, 1);   // global_prefetch_b8
    float acc = be[c];
#pragma unroll
    for (int k = 0; k < 16; ++k) acc = fmaf(row[k], We[k * O + c], acc);
    float msg = fmaxf(hsrc[s * O + c] + acc, 0.0f) + EPS_MSG;
    atomicMax(mbits + d * O + c, __float_as_uint(msg));
  }
}

// --------- edge pass 2: fused denom = Sum e^(m-M), numer = Sum m*e^(m-M) ----
__global__ void k_edge_agg(const float* __restrict__ ea,
                           const float* __restrict__ We,
                           const float* __restrict__ be,
                           const float* __restrict__ hsrc,
                           const long long* __restrict__ src,
                           const long long* __restrict__ dst,
                           const unsigned int* __restrict__ mbits,
                           float* __restrict__ denom, float* __restrict__ numer,
                           long long E, int shO) {
  const int       O     = 1 << shO;
  const long long total = E << shO;
  const long long step  = (long long)gridDim.x * blockDim.x;
  for (long long t = (long long)blockIdx.x * blockDim.x + threadIdx.x;
       t < total; t += step) {
    long long e = t >> shO;
    int       c = (int)(t & (O - 1));
    long long s = src[e], d = dst[e];
    const float* row = ea + e * 16;
    __builtin_prefetch(ea + (e + 256) * 16, 0, 1);
    float acc = be[c];
#pragma unroll
    for (int k = 0; k < 16; ++k) acc = fmaf(row[k], We[k * O + c], acc);
    float msg = fmaxf(hsrc[s * O + c] + acc, 0.0f) + EPS_MSG;
    float m   = __uint_as_float(mbits[d * O + c]);
    float ex  = __expf(msg - m);
    atomicAdd(denom + d * O + c, ex);
    atomicAdd(numer + d * O + c, msg * ex);
  }
}

// ------------- out = numer/(denom+eps) + h_dst  (softmax agg result) -------
__global__ void k_combine(const float* __restrict__ numer,
                          const float* __restrict__ denom,
                          const float* __restrict__ hdst,
                          float* __restrict__ out, long long total) {
  long long step = (long long)gridDim.x * blockDim.x;
  for (long long t = (long long)blockIdx.x * blockDim.x + threadIdx.x;
       t < total; t += step)
    out[t] = numer[t] / (denom[t] + EPS_SMX) + hdst[t];
}

// ---------------- BatchNorm batch statistics (one block / column) ----------
__global__ __launch_bounds__(256)
void k_bn_stats(const float* __restrict__ h, int Nrows, int Ccols,
                float* __restrict__ stats) {       // stats[c]=sum, stats[C+c]=sumsq
  const int c = blockIdx.x;
  float s = 0.0f, q = 0.0f;
  for (int i = threadIdx.x; i < Nrows; i += blockDim.x) {
    float v = h[(long long)i * Ccols + c];
    s += v; q += v * v;
  }
  __shared__ float sh[256], shq[256];
  sh[threadIdx.x] = s; shq[threadIdx.x] = q;
  __syncthreads();
  for (int o = 128; o > 0; o >>= 1) {
    if (threadIdx.x < o) {
      sh[threadIdx.x]  += sh[threadIdx.x + o];
      shq[threadIdx.x] += shq[threadIdx.x + o];
    }
    __syncthreads();
  }
  if (threadIdx.x == 0) { stats[c] = sh[0]; stats[Ccols + c] = shq[0]; }
}

__global__ void k_bn_apply(const float* __restrict__ in, float* __restrict__ out,
                           const float* __restrict__ stats,
                           const float* __restrict__ gamma,
                           const float* __restrict__ beta,
                           long long total, int shC, float invN, int relu) {
  const int C    = 1 << shC;
  long long step = (long long)gridDim.x * blockDim.x;
  for (long long t = (long long)blockIdx.x * blockDim.x + threadIdx.x;
       t < total; t += step) {
    int   c   = (int)(t & (C - 1));
    float mu  = stats[c] * invN;
    float var = stats[C + c] * invN - mu * mu;     // biased, matches jnp.var
    float y   = (in[t] - mu) * rsqrtf(var + EPS_BN) * gamma[c] + beta[c];
    if (relu) y = fmaxf(y, 0.0f);
    out[t] = y;
  }
}

// ----------------------------- mean pool -----------------------------------
__global__ void k_pool_accum(const float* __restrict__ h,
                             const long long* __restrict__ batch,
                             float* __restrict__ pool, float* __restrict__ cnt,
                             int Nrows, int shC) {
  const int C    = 1 << shC;
  long long tot  = (long long)Nrows << shC;
  long long step = (long long)gridDim.x * blockDim.x;
  for (long long t = (long long)blockIdx.x * blockDim.x + threadIdx.x;
       t < tot; t += step) {
    long long i = t >> shC;
    int       c = (int)(t & (C - 1));
    long long g = batch[i];
    atomicAdd(pool + (g << shC) + c, h[t]);
    if (c == 0) atomicAdd(cnt + g, 1.0f);
  }
}

__global__ void k_pool_final(const float* __restrict__ pool,
                             const float* __restrict__ cnt,
                             float* __restrict__ out, int shC) {
  int t = blockIdx.x * blockDim.x + threadIdx.x;
  if (t < (NUM_GRAPHS << shC))
    out[t] = pool[t] / fmaxf(cnt[t >> shC], 1.0f);
}

// ---------------------------------------------------------------------------
static inline int gs_blocks(long long total, int tpb) {
  long long b = (total + tpb - 1) / tpb;
  if (b > 16384) b = 16384;
  if (b < 1) b = 1;
  return (int)b;
}
static inline int ilog2(int v) { int s = 0; while ((1 << s) < v) ++s; return s; }

// pack W then run the WMMA GEMM (shares one packed-B scratch serially)
static void gemm(const float* A, const float* W, const float* bias, float* C,
                 int M, int K, int N, int relu, _Float16* packB,
                 hipStream_t stream) {
  int ptotal = (K / 32) * (N / 16) * 32 * 16;
  k_pack_b<<<(ptotal + 255) / 256, 256, 0, stream>>>(W, packB, K, N);
  dim3 blk(32, 4);
  dim3 grd(N / 16, (M / 16 + 3) / 4);
  k_gemm_wmma<<<grd, blk, 0, stream>>>(A, packB, bias, C, M, K, N, relu);
}

// One GENConv layer + outer BN/ReLU.  out must not alias S1..S5 or hin.
static void run_layer(const float* hin, int I, int O, const float* const* P,
                      const float* g_out, const float* b_out,
                      const float* ea, const long long* src, const long long* dst,
                      int Nn, long long E,
                      float* S1, float* S2, float* S3, float* S4, float* S5,
                      float* stats, _Float16* packB, float* out,
                      hipStream_t stream) {
  const float *Wsrc = P[0], *bsrc = P[1], *Wdst = P[2], *bdst = P[3];
  const float *Wedg = P[4], *bedg = P[5];
  const float *W1 = P[6], *b1 = P[7], *gam = P[8], *bet = P[9];
  const float *W2 = P[10], *b2 = P[11];

  const int O2  = 2 * O;
  const int shO = ilog2(O), shO2 = ilog2(O2);
  const long long NO  = (long long)Nn * O;
  const long long NO2 = (long long)Nn * O2;
  const float invN = 1.0f / (float)Nn;

  // node-side linears (WMMA)
  gemm(hin, Wsrc, bsrc, S1, Nn, I, O, 0, packB, stream);
  gemm(hin, Wdst, bdst, S2, Nn, I, O, 0, packB, stream);

  // init segment buffers (0.0f bits also serve as the positive-float max seed)
  k_fill<<<gs_blocks(NO, 256), 256, 0, stream>>>(S3, 0.0f, NO);
  k_fill<<<gs_blocks(NO, 256), 256, 0, stream>>>(S4, 0.0f, NO);
  k_fill<<<gs_blocks(NO, 256), 256, 0, stream>>>(S5, 0.0f, NO);

  // two fused edge passes (edge linear recomputed in-register each pass)
  k_edge_max<<<gs_blocks(E << shO, 256), 256, 0, stream>>>(
      ea, Wedg, bedg, S1, src, dst, (unsigned int*)S3, E, shO);
  k_edge_agg<<<gs_blocks(E << shO, 256), 256, 0, stream>>>(
      ea, Wedg, bedg, S1, src, dst, (const unsigned int*)S3, S4, S5, E, shO);

  // agg + h_dst  -> S1 (h_src dead now)
  k_combine<<<gs_blocks(NO, 256), 256, 0, stream>>>(S5, S4, S2, S1, NO);

  // MLP: S1 @ W1 -> hidden (S3||S4 contiguous N x 2O), BN+ReLU, @ W2 -> S2
  gemm(S1, W1, b1, S3, Nn, O, O2, 0, packB, stream);
  k_bn_stats<<<O2, 256, 0, stream>>>(S3, Nn, O2, stats);
  k_bn_apply<<<gs_blocks(NO2, 256), 256, 0, stream>>>(S3, S3, stats, gam, bet,
                                                      NO2, shO2, invN, 1);
  gemm(S3, W2, b2, S2, Nn, O2, O, 0, packB, stream);

  // outer BN + ReLU -> out
  k_bn_stats<<<O, 256, 0, stream>>>(S2, Nn, O, stats);
  k_bn_apply<<<gs_blocks(NO, 256), 256, 0, stream>>>(S2, out, stats, g_out, b_out,
                                                     NO, shO, invN, 1);
}

extern "C" void kernel_launch(void* const* d_in, const int* in_sizes, int n_in,
                              void* d_out, int out_size, void* d_ws, size_t ws_size,
                              hipStream_t stream) {
  // ---- inputs (setup_inputs dict order, leaves flattened in insertion order)
  const float*     x     = (const float*)d_in[0];
  const long long* eidx  = (const long long*)d_in[1];
  const float*     ea    = (const float*)d_in[2];
  const long long* batch = (const long long*)d_in[3];

  const int       Nn = in_sizes[3];          // 50000 nodes
  const long long E  = in_sizes[1] / 2;      // 800000 edges
  const int       I0 = in_sizes[0] / Nn;     // 64
  const int       O1 = in_sizes[5];          // lin_src_b of params1 -> 32
  const int       O2 = in_sizes[17];         // 64
  const int       O3 = in_sizes[29];         // 128

  const long long* src = eidx;
  const long long* dst = eidx + E;

  const float* P1[12]; const float* P2[12]; const float* P3[12];
  for (int i = 0; i < 12; ++i) {
    P1[i] = (const float*)d_in[4 + i];
    P2[i] = (const float*)d_in[16 + i];
    P3[i] = (const float*)d_in[28 + i];
  }
  const float* g1 = (const float*)d_in[40]; const float* b1 = (const float*)d_in[41];
  const float* g2 = (const float*)d_in[42]; const float* b2 = (const float*)d_in[43];
  const float* g3 = (const float*)d_in[44]; const float* b3 = (const float*)d_in[45];

  // ---- workspace carve-up (floats); planes of N x 128
  const long long P = (long long)Nn * 128;
  float* ws = (float*)d_ws;
  float* X0 = ws;               // layer-output ping
  float* X1 = ws + P;           // layer-output pong
  float* S1 = ws + 2 * P;       // h_src, then agg
  float* S2 = ws + 3 * P;       // h_dst, then raw layer out
  float* S3 = ws + 4 * P;       // max bits; S3||S4 = MLP hidden (N x 2O)
  float* S4 = ws + 5 * P;       // denom
  float* S5 = ws + 6 * P;       // numer
  float* stats = ws + 7 * P;    // 512 floats
  float* pool  = stats + 1024;  // 64*128 sums + 64 counts
  float* cnt   = pool + (long long)NUM_GRAPHS * 128;
  _Float16* packB = (_Float16*)(cnt + NUM_GRAPHS);  // <= 64KB packed weights

  // ---- 3 GENConv layers
  run_layer(x,  I0, O1, P1, g1, b1, ea, src, dst, Nn, E,
            S1, S2, S3, S4, S5, stats, packB, X0, stream);
  run_layer(X0, O1, O2, P2, g2, b2, ea, src, dst, Nn, E,
            S1, S2, S3, S4, S5, stats, packB, X1, stream);
  run_layer(X1, O2, O3, P3, g3, b3, ea, src, dst, Nn, E,
            S1, S2, S3, S4, S5, stats, packB, X0, stream);

  // ---- global mean pool over 64 graphs
  const int shC = ilog2(O3);
  const long long psz = ((long long)NUM_GRAPHS << shC) + NUM_GRAPHS;
  k_fill<<<gs_blocks(psz, 256), 256, 0, stream>>>(pool, 0.0f, psz);
  k_pool_accum<<<gs_blocks((long long)Nn << shC, 256), 256, 0, stream>>>(
      X0, batch, pool, cnt, Nn, shC);
  k_pool_final<<<(NUM_GRAPHS << shC) / 256, 256, 0, stream>>>(pool, cnt,
                                                             (float*)d_out, shC);
}
